// PerformerBiLSTMModel_17789754540045
// MI455X (gfx1250) — compile-verified, matrix operand order checked by reading, not run on hardware
//
#include <hip/hip_runtime.h>
#include <hip/hip_bf16.h>
#include <math.h>

// ---- model dims ----
#define BB 4
#define NN 512
#define IN_DIM 128
#define DD 256
#define HH 8
#define DH 32
#define LL 8
#define MM 110
#define MP 112          // padded feature count
#define HID 256
#define FF 1024
#define MR (BB*NN)      // 2048 token rows
#define BHN (BB*HH*NN)  // 16384 feature rows

typedef __attribute__((ext_vector_type(16))) _Float16 v16h;
typedef __attribute__((ext_vector_type(8)))  float    v8f;

union HalfPack { float4 f4; _Float16 h[8]; };
union FragU    { v16h v;  float4 f4[2]; };

// ---------------------------------------------------------------------------
// WMMA GEMM: C = act(A @ W^T + bias) (+ resid). A:[M,K] f32, W:[N,K] f32.
// Workgroup tile 128(M) x 64(N): 8 waves, each computes a 2x2 block of 16x16
// C-tiles (4 WMMAs per fragment-load group). LDS double-buffered: one barrier
// per 32-wide K step. f32 -> f16 conversion fused into the staging path.
// ---------------------------------------------------------------------------
template<int ACT>
__global__ __launch_bounds__(256) void wmma_gemm_kernel(
    const float* __restrict__ A, const float* __restrict__ W,
    const float* __restrict__ bias, const float* __restrict__ resid,
    float* __restrict__ C, int M, int N, int K)
{
    __shared__ __align__(16) _Float16 As[2][8][32][16];  // [buf][row-frag][lane][half]
    __shared__ __align__(16) _Float16 Bs[2][4][32][16];  // [buf][col-frag][lane][half]

    const int t  = threadIdx.x;
    const int m0 = blockIdx.y * 128;
    const int n0 = blockIdx.x * 64;

    const int wave = t >> 5;
    const int lane = t & 31;
    const int wr   = wave >> 1;          // 0..3 -> row frags 2wr, 2wr+1
    const int wc   = wave & 1;           // 0..1 -> col frags 2wc, 2wc+1
    const int fr0 = 2*wr, fr1 = 2*wr + 1;
    const int fc0 = 2*wc, fc1 = 2*wc + 1;

    v8f acc00 = {0.f,0.f,0.f,0.f,0.f,0.f,0.f,0.f};
    v8f acc01 = acc00, acc10 = acc00, acc11 = acc00;

    // stage one K-slab (32 wide) into LDS buffer `buf`.
    // A fragment 16x32 f16 layout: halves i<8 -> k=i+8*hi ; i>=8 -> k=i+8+8*hi
    auto stage = [&](int buf, int k0) {
        // B: 4 frags x 32 lanes x 2 groups = 256 -> one 8-half group per thread
        {
            const int frag = t >> 6;
            const int wi   = t & 63;
            const int l    = wi >> 1;
            const int i0   = (wi & 1) * 8;
            const int hi   = l >> 4;
            const int nr   = l & 15;
            const int kb   = (i0 == 0) ? (8*hi) : (16 + 8*hi);
            const float* p = W + (size_t)(n0 + frag*16 + nr) * K + k0 + kb;
            if (k0 + 32 < K) __builtin_prefetch(p + 32, 0, 1);   // global_prefetch_b8
            float4 b0v = *(const float4*)p;
            float4 b1v = *(const float4*)(p + 4);
            HalfPack pb;
            pb.h[0]=(_Float16)b0v.x; pb.h[1]=(_Float16)b0v.y; pb.h[2]=(_Float16)b0v.z; pb.h[3]=(_Float16)b0v.w;
            pb.h[4]=(_Float16)b1v.x; pb.h[5]=(_Float16)b1v.y; pb.h[6]=(_Float16)b1v.z; pb.h[7]=(_Float16)b1v.w;
            *(float4*)&Bs[buf][frag][l][i0] = pb.f4;
        }
        // A: 8 frags x 32 lanes x 2 groups = 512 -> two groups per thread
        #pragma unroll
        for (int g = 0; g < 2; ++g) {
            const int idx  = t + 256*g;
            const int frag = idx >> 6;
            const int wi   = idx & 63;
            const int l    = wi >> 1;
            const int i0   = (wi & 1) * 8;
            const int hi   = l >> 4;
            const int mr   = l & 15;
            const int kb   = (i0 == 0) ? (8*hi) : (16 + 8*hi);
            const float* p = A + (size_t)(m0 + frag*16 + mr) * K + k0 + kb;
            if (k0 + 32 < K) __builtin_prefetch(p + 32, 0, 1);
            float4 a0v = *(const float4*)p;
            float4 a1v = *(const float4*)(p + 4);
            HalfPack pa;
            pa.h[0]=(_Float16)a0v.x; pa.h[1]=(_Float16)a0v.y; pa.h[2]=(_Float16)a0v.z; pa.h[3]=(_Float16)a0v.w;
            pa.h[4]=(_Float16)a1v.x; pa.h[5]=(_Float16)a1v.y; pa.h[6]=(_Float16)a1v.z; pa.h[7]=(_Float16)a1v.w;
            *(float4*)&As[buf][frag][l][i0] = pa.f4;
        }
    };

    stage(0, 0);
    __syncthreads();
    int buf = 0;
    for (int k0 = 0; k0 < K; k0 += 32) {
        if (k0 + 32 < K) stage(buf ^ 1, k0 + 32);   // overlap staging with compute

        FragU fa0, fa1, fb0, fb1;
        fa0.f4[0] = *(const float4*)&As[buf][fr0][lane][0];
        fa0.f4[1] = *(const float4*)&As[buf][fr0][lane][8];
        fa1.f4[0] = *(const float4*)&As[buf][fr1][lane][0];
        fa1.f4[1] = *(const float4*)&As[buf][fr1][lane][8];
        fb0.f4[0] = *(const float4*)&Bs[buf][fc0][lane][0];
        fb0.f4[1] = *(const float4*)&Bs[buf][fc0][lane][8];
        fb1.f4[0] = *(const float4*)&Bs[buf][fc1][lane][0];
        fb1.f4[1] = *(const float4*)&Bs[buf][fc1][lane][8];
        acc00 = __builtin_amdgcn_wmma_f32_16x16x32_f16(false, fa0.v, false, fb0.v, (short)0, acc00, false, false);
        acc01 = __builtin_amdgcn_wmma_f32_16x16x32_f16(false, fa0.v, false, fb1.v, (short)0, acc01, false, false);
        acc10 = __builtin_amdgcn_wmma_f32_16x16x32_f16(false, fa1.v, false, fb0.v, (short)0, acc10, false, false);
        acc11 = __builtin_amdgcn_wmma_f32_16x16x32_f16(false, fa1.v, false, fb1.v, (short)0, acc11, false, false);

        __syncthreads();
        buf ^= 1;
    }

    // C/D layout: vgpr r -> M = r + 8*(lane>=16); N = lane&15
    const int hiL = lane >> 4;
    const int nl  = lane & 15;
    const int ng0 = n0 + fc0*16 + nl;
    const int ng1 = n0 + fc1*16 + nl;
    const float bv0 = bias[ng0];
    const float bv1 = bias[ng1];
    #pragma unroll
    for (int r = 0; r < 8; ++r) {
        const int mgA = m0 + fr0*16 + r + 8*hiL;
        const int mgB = m0 + fr1*16 + r + 8*hiL;
        float v00 = acc00[r] + bv0;
        float v01 = acc01[r] + bv1;
        float v10 = acc10[r] + bv0;
        float v11 = acc11[r] + bv1;
        if (ACT == 1) {
            v00 = 0.5f*v00*(1.0f + erff(v00*0.70710678f));
            v01 = 0.5f*v01*(1.0f + erff(v01*0.70710678f));
            v10 = 0.5f*v10*(1.0f + erff(v10*0.70710678f));
            v11 = 0.5f*v11*(1.0f + erff(v11*0.70710678f));
        }
        if (resid) {
            v00 += resid[(size_t)mgA * N + ng0];
            v01 += resid[(size_t)mgA * N + ng1];
            v10 += resid[(size_t)mgB * N + ng0];
            v11 += resid[(size_t)mgB * N + ng1];
        }
        C[(size_t)mgA * N + ng0] = v00;
        C[(size_t)mgA * N + ng1] = v01;
        C[(size_t)mgB * N + ng0] = v10;
        C[(size_t)mgB * N + ng1] = v11;
    }
}

// ---------------------------------------------------------------------------
// LayerNorm over D=256, one row per block.
// ---------------------------------------------------------------------------
__global__ __launch_bounds__(256) void layernorm_kernel(
    const float* __restrict__ x, const float* __restrict__ w,
    const float* __restrict__ b, float* __restrict__ y)
{
    __shared__ float red[256];
    const int t = threadIdx.x;
    const size_t row = blockIdx.x;
    float v = x[row*DD + t];
    red[t] = v; __syncthreads();
    for (int s = 128; s > 0; s >>= 1) { if (t < s) red[t] += red[t+s]; __syncthreads(); }
    float mu = red[0] * (1.0f/DD); __syncthreads();
    float d = v - mu;
    red[t] = d*d; __syncthreads();
    for (int s = 128; s > 0; s >>= 1) { if (t < s) red[t] += red[t+s]; __syncthreads(); }
    float var = red[0] * (1.0f/DD);
    y[row*DD + t] = d * rsqrtf(var + 1e-5f) * w[t] + b[t];
}

// ---------------------------------------------------------------------------
// FAVOR+ feature map. One block per (b,h,n) row, 128 threads.
// isQuery=1 : write final qp (per-row stabilizer)
// isQuery=0 : write raw dd + rowmax + diag (global stabilizer applied in pass2)
// ---------------------------------------------------------------------------
__global__ __launch_bounds__(128) void feat_kernel(
    const float* __restrict__ qk, const float* __restrict__ proj,
    float* __restrict__ outv, float* __restrict__ rowmax,
    float* __restrict__ diagbuf, int isQuery)
{
    __shared__ float xd[DH];
    __shared__ float red[128];
    __shared__ float sdiag, smax;
    const int t   = threadIdx.x;
    const int rid = blockIdx.x;            // rid = ((b*H+h)*N + n)
    const int n   = rid & (NN-1);
    const int bh  = rid >> 9;
    const int h   = bh & (HH-1);
    const int b   = bh >> 3;
    if (t < DH) xd[t] = qk[((size_t)(b*NN + n))*DD + h*DH + t] * 0.42044820762685725f; // DH^-0.25
    __syncthreads();
    red[t] = (t < DH) ? xd[t]*xd[t] : 0.f;
    __syncthreads();
    for (int s = 64; s > 0; s >>= 1) { if (t < s) red[t] += red[t+s]; __syncthreads(); }
    if (t == 0) sdiag = 0.5f * red[0];
    __syncthreads();
    float dd = -1e30f;
    if (t < MM) {
        const float* pr = proj + (size_t)t * DH;
        float s = 0.f;
        #pragma unroll
        for (int d = 0; d < DH; ++d) s += xd[d] * pr[d];
        dd = s;
    }
    red[t] = dd; __syncthreads();
    for (int s = 64; s > 0; s >>= 1) { if (t < s) red[t] = fmaxf(red[t], red[t+s]); __syncthreads(); }
    if (t == 0) smax = red[0];
    __syncthreads();
    if (isQuery) {
        if (t < MP)
            outv[(size_t)rid*MP + t] =
                (t < MM) ? 0.09534625892455922f * (expf(dd - sdiag - smax) + 1e-4f) : 0.f;
    } else {
        if (t < MP) outv[(size_t)rid*MP + t] = dd;
        if (t == 0) { rowmax[rid] = smax; diagbuf[rid] = sdiag; }
    }
}

// per-(b,h) max over N rows
__global__ __launch_bounds__(256) void kmax_kernel(
    const float* __restrict__ rowmax, float* __restrict__ kstab)
{
    __shared__ float red[256];
    const int bh = blockIdx.x, t = threadIdx.x;
    red[t] = fmaxf(rowmax[bh*NN + t], rowmax[bh*NN + 256 + t]);
    __syncthreads();
    for (int s = 128; s > 0; s >>= 1) { if (t < s) red[t] = fmaxf(red[t], red[t+s]); __syncthreads(); }
    if (t == 0) kstab[bh] = red[0];
}

__global__ void kfeat2_kernel(const float* __restrict__ dd, const float* __restrict__ diagk,
                              const float* __restrict__ kstab, float* __restrict__ kp, int total)
{
    int i = blockIdx.x*256 + threadIdx.x;
    if (i >= total) return;
    int m   = i % MP;
    int row = i / MP;
    int bh  = row >> 9;
    kp[i] = (m < MM) ? 0.09534625892455922f * (expf(dd[i] - diagk[row] - kstab[bh]) + 1e-4f) : 0.f;
}

// ---------------------------------------------------------------------------
// Causal linear attention. One block per (b,h); state kcum[MP], ctx[MP][DH] in LDS.
// ---------------------------------------------------------------------------
__global__ __launch_bounds__(256) void attn_kernel(
    const float* __restrict__ qp, const float* __restrict__ kp,
    const float* __restrict__ v, float* __restrict__ o)
{
    __shared__ float kcum[MP];
    __shared__ float ctx[MP*DH];
    __shared__ float xq[MP], xk[MP], vv[DH], red[256];
    __shared__ float sdinv;
    const int t  = threadIdx.x;
    const int bh = blockIdx.x;
    const int h  = bh & (HH-1), b = bh >> 3;
    for (int i = t; i < MP*DH; i += 256) ctx[i] = 0.f;
    if (t < MP) kcum[t] = 0.f;
    __syncthreads();
    for (int n = 0; n < NN; ++n) {
        const size_t frow = ((size_t)bh*NN + n)*MP;
        if (t < MP) { xq[t] = qp[frow + t]; xk[t] = kp[frow + t]; }
        if (t < DH) vv[t] = v[((size_t)(b*NN + n))*DD + h*DH + t];
        __syncthreads();
        if (t < MP) kcum[t] += xk[t];
        __syncthreads();
        red[t] = (t < MP) ? xq[t]*kcum[t] : 0.f;
        __syncthreads();
        for (int s = 128; s > 0; s >>= 1) { if (t < s) red[t] += red[t+s]; __syncthreads(); }
        if (t == 0) sdinv = 1.0f / (red[0] + 1e-6f);
        for (int i = t; i < MP*DH; i += 256) { int m = i >> 5, dh = i & 31; ctx[i] += xk[m]*vv[dh]; }
        __syncthreads();
        const int part = t >> 5, dh = t & 31;
        float p = 0.f;
        #pragma unroll
        for (int j = 0; j < 14; ++j) { int m = part*14 + j; p += ctx[m*DH + dh]*xq[m]; } // 8*14 = 112
        red[t] = p;
        __syncthreads();
        if (t < DH) {
            float s = 0.f;
            #pragma unroll
            for (int j = 0; j < 8; ++j) s += red[t + j*32];
            o[((size_t)(b*NN + n))*DD + h*DH + t] = s * sdinv;
        }
        __syncthreads();
    }
}

__global__ void add2_kernel(const float* __restrict__ a, const float* __restrict__ b,
                            float* __restrict__ c, int n)
{
    int i = blockIdx.x*256 + threadIdx.x;
    if (i < n) c[i] = a[i] + b[i];
}

// ---------------------------------------------------------------------------
// LSTM scan: blockIdx.x = direction (1 = reversed). gi precomputed by WMMA GEMM.
// h,c,[gates] live in LDS; recurrent matvec on VALU (latency-bound anyway).
// ---------------------------------------------------------------------------
__global__ __launch_bounds__(256) void lstm_scan_kernel(
    const float* __restrict__ gi, const float* __restrict__ Whh,
    float* __restrict__ out, int ldout)
{
    __shared__ float hsh[BB][HID];
    __shared__ float csh[BB][HID];
    __shared__ float gsh[BB][4*HID];
    const int t   = threadIdx.x;
    const int dir = blockIdx.x;
    const float* gi_d = gi  + (size_t)dir * MR * (4*HID);
    const float* whh  = Whh + (size_t)dir * (4*HID) * HID;
    const int colOff  = dir * HID;
    #pragma unroll
    for (int j = 0; j < 4; ++j) { int idx = t + 256*j; hsh[idx>>8][idx&255] = 0.f; csh[idx>>8][idx&255] = 0.f; }
    __syncthreads();
    for (int s = 0; s < NN; ++s) {
        const int n = dir ? (NN-1-s) : s;
        #pragma unroll 1
        for (int j = 0; j < 16; ++j) {
            int idx = t + 256*j;
            int bb = idx >> 10, jj = idx & 1023;
            float sum = gi_d[((size_t)(bb*NN + n))*(4*HID) + jj];
            const float4* wr = (const float4*)(whh + (size_t)jj*HID);
            const float* hb = hsh[bb];
            #pragma unroll 4
            for (int u = 0; u < HID/4; ++u) {
                float4 w4 = wr[u];
                sum += w4.x*hb[4*u] + w4.y*hb[4*u+1] + w4.z*hb[4*u+2] + w4.w*hb[4*u+3];
            }
            gsh[bb][jj] = sum;
        }
        __syncthreads();
        #pragma unroll
        for (int j = 0; j < 4; ++j) {
            int idx = t + 256*j;
            int bb = idx >> 8, u = idx & 255;
            float ig = 1.f/(1.f + expf(-gsh[bb][u]));
            float fg = 1.f/(1.f + expf(-gsh[bb][HID + u]));
            float gg = tanhf(gsh[bb][2*HID + u]);
            float og = 1.f/(1.f + expf(-gsh[bb][3*HID + u]));
            float c  = fg*csh[bb][u] + ig*gg;
            csh[bb][u] = c;
            float hv = og*tanhf(c);
            hsh[bb][u] = hv;
            out[((size_t)(bb*NN + n))*ldout + colOff + u] = hv;
        }
        __syncthreads();
    }
}

// final head: out[b] = dot(h[b, N-1, :512], fc_w) + fc_b
__global__ __launch_bounds__(128) void head_kernel(
    const float* __restrict__ cc, const float* __restrict__ fcw,
    const float* __restrict__ fcb, float* __restrict__ out)
{
    __shared__ float red[128];
    const int t = threadIdx.x, b = t >> 5, lane = t & 31;
    const float* hr = cc + ((size_t)(b*NN + NN-1)) * (2*HID);
    float p = 0.f;
    for (int j = lane; j < 2*HID; j += 32) p += hr[j]*fcw[j];
    red[t] = p; __syncthreads();
    for (int s = 16; s > 0; s >>= 1) { if (lane < s) red[t] += red[t+s]; __syncthreads(); }
    if (lane == 0) out[b] = red[b*32] + fcb[0];
}

// ---------------------------------------------------------------------------
extern "C" void kernel_launch(void* const* d_in, const int* in_sizes, int n_in,
                              void* d_out, int out_size, void* d_ws, size_t ws_size,
                              hipStream_t stream) {
    const float* X     = (const float*)d_in[0];
    const float* Win_w = (const float*)d_in[1];
    const float* Win_b = (const float*)d_in[2];
    const float* ln1_w = (const float*)d_in[3];
    const float* ln1_b = (const float*)d_in[4];
    const float* Wq    = (const float*)d_in[5];
    const float* bq    = (const float*)d_in[6];
    const float* Wk    = (const float*)d_in[7];
    const float* bk    = (const float*)d_in[8];
    const float* Wv    = (const float*)d_in[9];
    const float* bv    = (const float*)d_in[10];
    const float* Wo    = (const float*)d_in[11];
    const float* bo    = (const float*)d_in[12];
    const float* proj  = (const float*)d_in[13];
    const float* ln2_w = (const float*)d_in[14];
    const float* ln2_b = (const float*)d_in[15];
    const float* Wff1  = (const float*)d_in[16];
    const float* bff1  = (const float*)d_in[17];
    const float* Wff2  = (const float*)d_in[18];
    const float* bff2  = (const float*)d_in[19];
    const float* Wih0  = (const float*)d_in[20];
    const float* Whh0  = (const float*)d_in[21];
    const float* b0    = (const float*)d_in[22];
    const float* Wih12 = (const float*)d_in[23];
    const float* Whh12 = (const float*)d_in[24];
    const float* b12   = (const float*)d_in[25];
    const float* fc_w  = (const float*)d_in[26];
    const float* fc_b  = (const float*)d_in[27];
    float* out = (float*)d_out;

    // workspace layout (f32)
    float* w0 = (float*)d_ws;
    size_t off = 0;
    auto alloc = [&](size_t n) { float* p = w0 + off; off += (n + 63) & ~(size_t)63; return p; };
    float* hbuf   = alloc((size_t)MR*DD);
    float* ybuf   = alloc((size_t)MR*DD);
    float* qbuf   = alloc((size_t)MR*DD);
    float* kbuf   = alloc((size_t)MR*DD);
    float* vbuf   = alloc((size_t)MR*DD);
    float* obuf   = alloc((size_t)MR*DD);
    float* bigbuf = alloc((size_t)2*MR*1024);   // reused: ddk / FF intermediate / LSTM gi (both dirs)
    float* qp     = alloc((size_t)BHN*MP);
    float* kp     = alloc((size_t)BHN*MP);
    float* rowmax = alloc((size_t)BHN);
    float* diagk  = alloc((size_t)BHN);
    float* kstab  = alloc((size_t)BB*HH);
    float* cc0    = alloc((size_t)MR*2*HID);
    float* cc1    = alloc((size_t)MR*2*HID);
    float* bcomb  = alloc((size_t)2*1024);

    auto gemm = [&](const float* A, const float* W, const float* bias, const float* resid,
                    float* C, int M_, int N_, int K_, int act) {
        dim3 g(N_/64, M_/128);
        if (act == 1)
            wmma_gemm_kernel<1><<<g, 256, 0, stream>>>(A, W, bias, resid, C, M_, N_, K_);
        else
            wmma_gemm_kernel<0><<<g, 256, 0, stream>>>(A, W, bias, resid, C, M_, N_, K_);
    };

    // input projection: [2048,128] @ Win^T -> h [2048,256]
    gemm(X, Win_w, Win_b, nullptr, hbuf, MR, DD, IN_DIM, 0);

    // ---- 8 performer blocks ----
    for (int l = 0; l < LL; ++l) {
        layernorm_kernel<<<MR, 256, 0, stream>>>(hbuf, ln1_w + l*DD, ln1_b + l*DD, ybuf);
        gemm(ybuf, Wq + (size_t)l*DD*DD, bq + l*DD, nullptr, qbuf, MR, DD, DD, 0);
        gemm(ybuf, Wk + (size_t)l*DD*DD, bk + l*DD, nullptr, kbuf, MR, DD, DD, 0);
        gemm(ybuf, Wv + (size_t)l*DD*DD, bv + l*DD, nullptr, vbuf, MR, DD, DD, 0);
        const float* pr = proj + (size_t)l*MM*DH;
        feat_kernel<<<BHN, 128, 0, stream>>>(qbuf, pr, qp, nullptr, nullptr, 1);
        feat_kernel<<<BHN, 128, 0, stream>>>(kbuf, pr, bigbuf, rowmax, diagk, 0);
        kmax_kernel<<<BB*HH, 256, 0, stream>>>(rowmax, kstab);
        kfeat2_kernel<<<(BHN*MP)/256, 256, 0, stream>>>(bigbuf, diagk, kstab, kp, BHN*MP);
        attn_kernel<<<BB*HH, 256, 0, stream>>>(qp, kp, vbuf, obuf);
        gemm(obuf, Wo + (size_t)l*DD*DD, bo + l*DD, hbuf, hbuf, MR, DD, DD, 0);  // += residual
        layernorm_kernel<<<MR, 256, 0, stream>>>(hbuf, ln2_w + l*DD, ln2_b + l*DD, ybuf);
        gemm(ybuf, Wff1 + (size_t)l*FF*DD, bff1 + l*FF, nullptr, bigbuf, MR, FF, DD, 1); // gelu
        gemm(bigbuf, Wff2 + (size_t)l*DD*FF, bff2 + l*DD, hbuf, hbuf, MR, DD, FF, 0);    // += residual
    }

    // ---- BiLSTM layer 0 (input D=256) ----
    for (int d = 0; d < 2; ++d) {
        add2_kernel<<<4, 256, 0, stream>>>(b0 + (size_t)d*2*1024, b0 + (size_t)d*2*1024 + 1024,
                                           bcomb + d*1024, 1024);
        gemm(hbuf, Wih0 + (size_t)d*1024*DD, bcomb + d*1024, nullptr,
             bigbuf + (size_t)d*MR*1024, MR, 1024, DD, 0);
    }
    lstm_scan_kernel<<<2, 256, 0, stream>>>(bigbuf, Whh0, cc0, 2*HID);

    // ---- BiLSTM layers 1,2 (input 2*HID=512) ----
    float* inbuf = cc0;
    float* outbuf = cc1;
    for (int lay = 0; lay < 2; ++lay) {
        for (int d = 0; d < 2; ++d) {
            const float* bsrc = b12 + ((size_t)(lay*2 + d)*2)*1024;
            add2_kernel<<<4, 256, 0, stream>>>(bsrc, bsrc + 1024, bcomb + d*1024, 1024);
            gemm(inbuf, Wih12 + (size_t)(lay*2 + d)*1024*(2*HID), bcomb + d*1024, nullptr,
                 bigbuf + (size_t)d*MR*1024, MR, 1024, 2*HID, 0);
        }
        lstm_scan_kernel<<<2, 256, 0, stream>>>(bigbuf, Whh12 + (size_t)lay*2*1024*HID,
                                                outbuf, 2*HID);
        float* tmp = inbuf; inbuf = outbuf; outbuf = tmp;
    }

    head_kernel<<<1, 128, 0, stream>>>(inbuf, fc_w, fc_b, out);
}
